// VarAttentionBlock_81793357185630
// MI455X (gfx1250) — matrix operand order repaired and checked by reading, not compiled, not run on hardware
//
#include <hip/hip_runtime.h>
#include <hip/hip_bf16.h>

// ---------------- problem constants ----------------
#define B_   8
#define V_   64
#define T_   256
#define T1_  257
#define D_   256
#define H_   8
#define DH_  32
#define ROWS_ (B_*V_*T1_)   // 131584 tokens, divisible by 128

typedef __bf16 v16bf __attribute__((ext_vector_type(16)));
typedef float  v8f   __attribute__((ext_vector_type(8)));

union FragAB { v16bf v; uint4 q[2]; unsigned short h[16]; };
union Pack8  { unsigned short s[8]; uint4 q; };

__device__ __forceinline__ unsigned short f2bf(float f) {
  unsigned int u = __float_as_uint(f);
  u += 0x7FFFu + ((u >> 16) & 1u);     // round-to-nearest-even
  return (unsigned short)(u >> 16);
}

__device__ __forceinline__ v8f wmma_bf16(v16bf a, v16bf b, v8f c) {
  // D = A(16x32 bf16) * B(32x16 bf16) + C(16x16 f32)
  return __builtin_amdgcn_wmma_f32_16x16x32_bf16(false, a, false, b, (short)0, c,
                                                 false, false);
}

// A fragment (16x32 bf16): lane half=lane>>4 picks K {0..15} or
// {16..31} -> two 16B chunks at ushort offsets half*16 and half*16+8.
__device__ __forceinline__ v16bf load_a_frag(const unsigned short* rowPtr, int half) {
  FragAB f;
  f.q[0] = *(const uint4*)(rowPtr + half * 16);
  f.q[1] = *(const uint4*)(rowPtr + half * 16 + 8);
  return f.v;
}
// B fragment (32x16 bf16) from (N,K) row-major weights: column n = lane&15,
// k = half*16 + j -> 16 consecutive bf16.
__device__ __forceinline__ v16bf load_b_frag(const unsigned short* p) {
  FragAB f;
  f.q[0] = *(const uint4*)(p);
  f.q[1] = *(const uint4*)(p + 8);
  return f.v;
}

// ---------------- K0: weights f32 -> bf16 ----------------
__global__ __launch_bounds__(256) void k_convert_w(const float* __restrict__ wqkv,
                                                   const float* __restrict__ wproj,
                                                   unsigned short* __restrict__ wq_bf,
                                                   unsigned short* __restrict__ wp_bf) {
  int i = blockIdx.x * 256 + threadIdx.x;
  if (i < 3 * D_ * D_) wq_bf[i] = f2bf(wqkv[i]);
  if (i < D_ * D_)     wp_bf[i] = f2bf(wproj[i]);
}

// ---------------- K1: LayerNorm + masked temporal mean ----------------
// One workgroup per (b,v); one wave per token; lane owns 8 channels.
__global__ __launch_bounds__(256) void k_ln(const float* __restrict__ x,
                                            const unsigned char* __restrict__ obs,
                                            unsigned short* __restrict__ xn,
                                            float* __restrict__ xnm,
                                            const float* __restrict__ lnw,
                                            const float* __restrict__ lnb) {
  __shared__ float acc[D_];
  __shared__ int cnt;
  const int tid = threadIdx.x, bv = blockIdx.x;
  acc[tid] = 0.f;
  if (tid == 0) cnt = 0;
  __syncthreads();
  const int wave = tid >> 5, lane = tid & 31;
  float wv8[8], bv8[8];
#pragma unroll
  for (int j = 0; j < 8; ++j) { wv8[j] = lnw[lane * 8 + j]; bv8[j] = lnb[lane * 8 + j]; }

  for (int t = wave; t < T1_; t += 8) {
    const size_t base = ((size_t)bv * T1_ + t) * D_ + lane * 8;
    float xv[8];
    float4 a0 = *(const float4*)(x + base);
    float4 a1 = *(const float4*)(x + base + 4);
    xv[0]=a0.x; xv[1]=a0.y; xv[2]=a0.z; xv[3]=a0.w;
    xv[4]=a1.x; xv[5]=a1.y; xv[6]=a1.z; xv[7]=a1.w;
    float s = 0.f, ss = 0.f;
#pragma unroll
    for (int j = 0; j < 8; ++j) { s += xv[j]; ss += xv[j] * xv[j]; }
#pragma unroll
    for (int off = 16; off; off >>= 1) {
      s  += __shfl_xor(s,  off, 32);
      ss += __shfl_xor(ss, off, 32);
    }
    const float mu = s * (1.f / D_);
    const float rs = rsqrtf(ss * (1.f / D_) - mu * mu + 1e-5f);
    float y[8];
    Pack8 pk;
#pragma unroll
    for (int j = 0; j < 8; ++j) {
      y[j] = (xv[j] - mu) * rs * wv8[j] + bv8[j];
      pk.s[j] = f2bf(y[j]);
    }
    *(uint4*)(xn + base) = pk.q;                 // bf16 xn, same flat layout
    const bool m = (t == 0) || (obs[(size_t)bv * T_ + (t - 1)] != 0);
    if (m) {
#pragma unroll
      for (int j = 0; j < 8; ++j) atomicAdd(&acc[lane * 8 + j], y[j]);  // ds_add_f32
      if (lane == 0) atomicAdd(&cnt, 1);
    }
  }
  __syncthreads();
  xnm[(size_t)bv * D_ + tid] = acc[tid] * (1.0f / (float)cnt);  // masked mean of xn
}

// ---------------- K2: q,k,scores,softmax -> attn (bf16) ----------------
// One workgroup per b; wave h owns head h entirely. Softmax done with
// cross-lane shuffles directly on WMMA accumulator fragments.
__global__ __launch_bounds__(256) void k_attn(const unsigned short* __restrict__ xn,
                                              const float* __restrict__ xnm,
                                              const unsigned short* __restrict__ wq_bf,
                                              unsigned short* __restrict__ attn) {
  __shared__ __align__(16) unsigned short s_q[H_ * V_ * DH_];  // 32 KB
  __shared__ __align__(16) unsigned short s_k[H_ * V_ * DH_];  // 32 KB
  const int b = blockIdx.x;
  const int tid = threadIdx.x, lane = tid & 31, h = tid >> 5;
  const int half = lane >> 4, ln = lane & 15;
  const float qscale = 0.17677669529663687f;   // dh^-0.5, folded into q

  // --- q_h = xn[:, t=0, :] @ Wq_h^T ---
  {
    v8f c[4][2] = {};
#pragma unroll
    for (int kb = 0; kb < 8; ++kb) {
      v16bf afr[4], bfr[2];
#pragma unroll
      for (int mt = 0; mt < 4; ++mt) {
        const int v = mt * 16 + ln;
        afr[mt] = load_a_frag(xn + ((size_t)(b * V_ + v) * T1_) * D_ + kb * 32, half);
      }
#pragma unroll
      for (int nt = 0; nt < 2; ++nt) {
        const int n = h * DH_ + nt * 16 + ln;       // q rows 0..255 of w_qkv
        bfr[nt] = load_b_frag(wq_bf + (size_t)n * D_ + kb * 32 + half * 16);
      }
#pragma unroll
      for (int mt = 0; mt < 4; ++mt)
#pragma unroll
        for (int nt = 0; nt < 2; ++nt) c[mt][nt] = wmma_bf16(afr[mt], bfr[nt], c[mt][nt]);
    }
#pragma unroll
    for (int mt = 0; mt < 4; ++mt)
#pragma unroll
      for (int nt = 0; nt < 2; ++nt)
#pragma unroll
        for (int r = 0; r < 8; ++r)
          s_q[(h * V_ + mt * 16 + r + 8 * half) * DH_ + nt * 16 + ln] =
              f2bf(c[mt][nt][r] * qscale);
  }
  // --- k_h = mean_masked(xn) @ Wk_h^T (mean precomputed in f32) ---
  {
    v8f c[4][2] = {};
#pragma unroll
    for (int kb = 0; kb < 8; ++kb) {
      v16bf afr[4], bfr[2];
#pragma unroll
      for (int mt = 0; mt < 4; ++mt) {
        FragAB f;
        const int v = mt * 16 + ln;
        const float* p = xnm + (size_t)(b * V_ + v) * D_ + kb * 32;
#pragma unroll
        for (int j = 0; j < 16; ++j) f.h[j] = f2bf(p[j + 16 * half]);
        afr[mt] = f.v;
      }
#pragma unroll
      for (int nt = 0; nt < 2; ++nt) {
        const int n = D_ + h * DH_ + nt * 16 + ln;  // k rows 256..511 of w_qkv
        bfr[nt] = load_b_frag(wq_bf + (size_t)n * D_ + kb * 32 + half * 16);
      }
#pragma unroll
      for (int mt = 0; mt < 4; ++mt)
#pragma unroll
        for (int nt = 0; nt < 2; ++nt) c[mt][nt] = wmma_bf16(afr[mt], bfr[nt], c[mt][nt]);
    }
#pragma unroll
    for (int mt = 0; mt < 4; ++mt)
#pragma unroll
      for (int nt = 0; nt < 2; ++nt)
#pragma unroll
        for (int r = 0; r < 8; ++r)
          s_k[(h * V_ + mt * 16 + r + 8 * half) * DH_ + nt * 16 + ln] =
              f2bf(c[mt][nt][r]);
  }
  __syncthreads();
  // --- S = q @ k^T (K = DH = one WMMA step), softmax rows, store attn ---
#pragma unroll
  for (int mt = 0; mt < 4; ++mt) {
    const v16bf a = load_a_frag(s_q + (size_t)(h * V_ + mt * 16 + ln) * DH_, half);
    v8f c[4];
#pragma unroll
    for (int nt = 0; nt < 4; ++nt) {
      const int vp = nt * 16 + ln;
      v8f z = {};
      c[nt] = wmma_bf16(a, load_b_frag(s_k + (size_t)(h * V_ + vp) * DH_ + half * 16), z);
    }
#pragma unroll
    for (int r = 0; r < 8; ++r) {
      float m = fmaxf(fmaxf(c[0][r], c[1][r]), fmaxf(c[2][r], c[3][r]));
#pragma unroll
      for (int off = 8; off; off >>= 1) m = fmaxf(m, __shfl_xor(m, off, 32));
      float e[4], s = 0.f;
#pragma unroll
      for (int nt = 0; nt < 4; ++nt) { e[nt] = __expf(c[nt][r] - m); s += e[nt]; }
#pragma unroll
      for (int off = 8; off; off >>= 1) s += __shfl_xor(s, off, 32);
      const float inv = 1.f / s;
      const int row = mt * 16 + r + 8 * half;
#pragma unroll
      for (int nt = 0; nt < 4; ++nt)
        attn[((size_t)(b * H_ + h) * V_ + row) * V_ + nt * 16 + ln] = f2bf(e[nt] * inv);
    }
  }
}

// ---------------- K3: vf = xn @ Wv^T (bf16 WMMA GEMM) ----------------
// WG = 128x128 tile, 8 waves (4 row x 2 col), wave = 32x64. A and B fragments
// come straight from global: xn (67MB) + weights live in the 192MB L2.
__global__ __launch_bounds__(256) void k_gemm_vf(const unsigned short* __restrict__ xn,
                                                 const unsigned short* __restrict__ wv,
                                                 unsigned short* __restrict__ vf) {
  const int tid = threadIdx.x, lane = tid & 31, w = tid >> 5;
  const int half = lane >> 4, ln = lane & 15;
  const int rowbase = blockIdx.x * 128 + (w >> 1) * 32;
  const int colbase = blockIdx.y * 128 + (w & 1) * 64;
  v8f c[2][4] = {};
#pragma unroll
  for (int kb = 0; kb < 8; ++kb) {
    v16bf a[2], bb[4];
#pragma unroll
    for (int mt = 0; mt < 2; ++mt) {
      const unsigned short* p = xn + (size_t)(rowbase + mt * 16 + ln) * D_ + kb * 32;
      a[mt] = load_a_frag(p, half);
      if (kb < 7) __builtin_prefetch(p + 32, 0, 1);    // global_prefetch next k-block
    }
#pragma unroll
    for (int nt = 0; nt < 4; ++nt)
      bb[nt] = load_b_frag(wv + (size_t)(colbase + nt * 16 + ln) * D_ + kb * 32 + half * 16);
#pragma unroll
    for (int mt = 0; mt < 2; ++mt)
#pragma unroll
      for (int nt = 0; nt < 4; ++nt) c[mt][nt] = wmma_bf16(a[mt], bb[nt], c[mt][nt]);
  }
#pragma unroll
  for (int mt = 0; mt < 2; ++mt)
#pragma unroll
    for (int nt = 0; nt < 4; ++nt)
#pragma unroll
      for (int r = 0; r < 8; ++r) {
        const int row = rowbase + mt * 16 + r + 8 * half;
        const int col = colbase + nt * 16 + ln;
        vf[(size_t)row * D_ + col] = f2bf(c[mt][nt][r]);
      }
}

// ---------------- K4: per-(b,t,h) value mixing: op_h = attn_h @ vf_h ---------
// vf head-blocks are staged global->LDS with CDNA5 async-to-LDS DMA
// (ASYNCcnt-tracked, no VGPR round trip), then consumed by WMMA.
__global__ __launch_bounds__(256) void k_mix(const unsigned short* __restrict__ vf,
                                             const unsigned short* __restrict__ attn,
                                             unsigned short* __restrict__ op) {
  __shared__ __align__(16) unsigned short sV[H_][V_][DH_];   // 32 KB
  const int t = blockIdx.x, b = blockIdx.y;
  const int tid = threadIdx.x, lane = tid & 31, h = tid >> 5;
  const int half = lane >> 4, ln = lane & 15;
  // stage this head's 64x32 vf block: each lane DMAs two 64B rows to LDS.
  // INST_OFFSET is added to BOTH the LDS and global addresses (ISA 10.async),
  // so one base pair + offsets covers the row.
#pragma unroll
  for (int rr = 0; rr < 2; ++rr) {
    const int v = lane + rr * 32;
    const unsigned int ldsa =
        (unsigned int)(uintptr_t)&sV[h][v][0];   // low 32 bits = LDS offset
    const unsigned long long ga = (unsigned long long)(uintptr_t)(
        vf + (((size_t)(b * V_ + v) * T1_ + t) * D_) + h * DH_);
    asm volatile(
        "global_load_async_to_lds_b128 %0, %1, off\n\t"
        "global_load_async_to_lds_b128 %0, %1, off offset:16\n\t"
        "global_load_async_to_lds_b128 %0, %1, off offset:32\n\t"
        "global_load_async_to_lds_b128 %0, %1, off offset:48"
        :: "v"(ldsa), "v"(ga) : "memory");
  }
  asm volatile("s_wait_asynccnt 0" ::: "memory");  // wave h consumes its own block
  __syncthreads();
  v8f c[4][2] = {};
#pragma unroll
  for (int kb = 0; kb < 2; ++kb) {
    v16bf a[4], bb[2];
#pragma unroll
    for (int mt = 0; mt < 4; ++mt)
      a[mt] = load_a_frag(attn + ((size_t)(b * H_ + h) * V_ + mt * 16 + ln) * V_ + kb * 32, half);
#pragma unroll
    for (int nt = 0; nt < 2; ++nt) {
      FragAB f;
      const int n = nt * 16 + ln;
#pragma unroll
      for (int j = 0; j < 16; ++j) f.h[j] = sV[h][kb * 32 + half * 16 + j][n];
      bb[nt] = f.v;
    }
#pragma unroll
    for (int mt = 0; mt < 4; ++mt)
#pragma unroll
      for (int nt = 0; nt < 2; ++nt) c[mt][nt] = wmma_bf16(a[mt], bb[nt], c[mt][nt]);
  }
#pragma unroll
  for (int mt = 0; mt < 4; ++mt)
#pragma unroll
    for (int nt = 0; nt < 2; ++nt)
#pragma unroll
      for (int r = 0; r < 8; ++r) {
        const int q = mt * 16 + r + 8 * half;
        op[(((size_t)(b * V_ + q) * T1_ + t) * D_) + h * DH_ + nt * 16 + ln] =
            f2bf(c[mt][nt][r]);
      }
}

// ---------------- K5: out = x + op @ Wp^T + b_proj (f32 epilogue) ----------
__global__ __launch_bounds__(256) void k_proj(const unsigned short* __restrict__ op,
                                              const unsigned short* __restrict__ wp,
                                              const float* __restrict__ x,
                                              const float* __restrict__ bproj,
                                              float* __restrict__ out) {
  const int tid = threadIdx.x, lane = tid & 31, w = tid >> 5;
  const int half = lane >> 4, ln = lane & 15;
  const int rowbase = blockIdx.x * 128 + (w >> 1) * 32;
  const int colbase = blockIdx.y * 128 + (w & 1) * 64;
  v8f c[2][4] = {};
#pragma unroll
  for (int kb = 0; kb < 8; ++kb) {
    v16bf a[2], bb[4];
#pragma unroll
    for (int mt = 0; mt < 2; ++mt) {
      const unsigned short* p = op + (size_t)(rowbase + mt * 16 + ln) * D_ + kb * 32;
      a[mt] = load_a_frag(p, half);
      if (kb < 7) __builtin_prefetch(p + 32, 0, 1);
    }
#pragma unroll
    for (int nt = 0; nt < 4; ++nt)
      bb[nt] = load_b_frag(wp + (size_t)(colbase + nt * 16 + ln) * D_ + kb * 32 + half * 16);
#pragma unroll
    for (int mt = 0; mt < 2; ++mt)
#pragma unroll
      for (int nt = 0; nt < 4; ++nt) c[mt][nt] = wmma_bf16(a[mt], bb[nt], c[mt][nt]);
  }
#pragma unroll
  for (int nt = 0; nt < 4; ++nt) {
    const int col = colbase + nt * 16 + ln;
    const float bp = bproj[col];
#pragma unroll
    for (int mt = 0; mt < 2; ++mt)
#pragma unroll
      for (int r = 0; r < 8; ++r) {
        const size_t idx = (size_t)(rowbase + mt * 16 + r + 8 * half) * D_ + col;
        out[idx] = c[mt][nt][r] + bp + x[idx];
      }
  }
}

// ---------------- host launcher ----------------
extern "C" void kernel_launch(void* const* d_in, const int* in_sizes, int n_in,
                              void* d_out, int out_size, void* d_ws, size_t ws_size,
                              hipStream_t stream) {
  const float*         x     = (const float*)d_in[0];
  const unsigned char* obs   = (const unsigned char*)d_in[1];  // jnp.bool_ = 1 byte
  const float*         lnw   = (const float*)d_in[2];
  const float*         lnb   = (const float*)d_in[3];
  const float*         wqkv  = (const float*)d_in[4];
  const float*         wproj = (const float*)d_in[5];
  const float*         bproj = (const float*)d_in[6];
  float* out = (float*)d_out;

  char* ws = (char*)d_ws;
  const size_t offXn  = 0;                                       // xn bf16, reused as op
  const size_t offVf  = offXn  + (size_t)ROWS_ * D_ * 2;         // vf bf16
  const size_t offXnm = offVf  + (size_t)ROWS_ * D_ * 2;         // masked mean, f32
  const size_t offWq  = offXnm + (size_t)B_ * V_ * D_ * 4;       // w_qkv bf16
  const size_t offWp  = offWq  + (size_t)3 * D_ * D_ * 2;        // w_proj bf16
  const size_t offAt  = offWp  + (size_t)D_ * D_ * 2;            // attn bf16

  unsigned short* xn    = (unsigned short*)(ws + offXn);
  unsigned short* vf    = (unsigned short*)(ws + offVf);
  float*          xnm   = (float*)(ws + offXnm);
  unsigned short* wq_bf = (unsigned short*)(ws + offWq);
  unsigned short* wp_bf = (unsigned short*)(ws + offWp);
  unsigned short* attn  = (unsigned short*)(ws + offAt);

  k_convert_w<<<768, 256, 0, stream>>>(wqkv, wproj, wq_bf, wp_bf);
  k_ln<<<B_ * V_, 256, 0, stream>>>(x, obs, xn, xnm, lnw, lnb);
  k_attn<<<B_, 256, 0, stream>>>(xn, xnm, wq_bf, attn);
  k_gemm_vf<<<dim3(ROWS_ / 128, 2), 256, 0, stream>>>(xn, wq_bf + (size_t)2 * D_ * D_, vf);
  k_mix<<<dim3(T1_, B_), 256, 0, stream>>>(vf, attn, xn /* reuse as op */);
  k_proj<<<dim3(ROWS_ / 128, 2), 256, 0, stream>>>(xn /* op */, wp_bf, x, bproj, out);
}
